// RNN_19155554140953
// MI455X (gfx1250) — matrix operand order repaired and checked by reading, not compiled, not run on hardware
//
#include <hip/hip_runtime.h>
#include <hip/hip_bf16.h>
#include <math.h>

// B=64, S=512, I=256, H=512, L=2, C=10
#define BB 64
#define SS 512
#define II 256
#define HH 512
#define CC 10
#define MM (BB * SS)

typedef __attribute__((ext_vector_type(16))) __bf16 bf16x16;
typedef __attribute__((ext_vector_type(8)))  float  f32x8;
typedef __attribute__((ext_vector_type(4))) unsigned u32x4;
typedef __attribute__((ext_vector_type(8))) unsigned u32x8;

// ---------- helpers ----------
__device__ __forceinline__ unsigned short f2bf(float f) {
  unsigned u = __builtin_bit_cast(unsigned, f);
  unsigned r = u + 0x7FFFu + ((u >> 16) & 1u);   // RNE
  return (unsigned short)(r >> 16);
}
__device__ __forceinline__ __bf16 us2bf(unsigned short h) {
  return __builtin_bit_cast(__bf16, h);
}
__device__ __forceinline__ f32x8 wmma_bf16(bf16x16 a, bf16x16 b, f32x8 c) {
  return __builtin_amdgcn_wmma_f32_16x16x32_bf16(false, a, false, b, (short)0, c,
                                                 false, false);
}
// CDNA5 async DMA: per-lane 16B global -> LDS, tracked by ASYNCcnt.
__device__ __forceinline__ void async_b128(unsigned lds_addr, const void* gaddr) {
  asm volatile("global_load_async_to_lds_b128 %0, %1, off"
               :: "v"(lds_addr), "v"((unsigned long long)(size_t)gaddr)
               : "memory");
}
__device__ __forceinline__ void wait_async_0() {
  asm volatile("s_wait_asynccnt 0x0" ::: "memory");
}
__device__ __forceinline__ void wait_async_3() {
  asm volatile("s_wait_asynccnt 0x3" ::: "memory");
}
// CDNA5 transcendental tanh (TRANS32). v_nop covers the TRANS result hazard.
__device__ __forceinline__ float fast_tanh(float x) {
  float r;
  asm volatile("v_tanh_f32 %0, %1\n\tv_nop" : "=v"(r) : "v"(x));
  return r;
}
// CDNA5 Tensor Data Mover: 2-D strided tile (global) -> packed LDS.
// D# bitfields per CDNA5 ISA.
__device__ __forceinline__ void tdm_load_2d(unsigned lds_addr, const void* gaddr,
                                            unsigned tensor_d0, unsigned tensor_d1,
                                            unsigned tile_d0, unsigned tile_d1,
                                            unsigned stride_d0 /*elements*/) {
  unsigned long long ga = (unsigned long long)(size_t)gaddr;
  u32x4 g0;
  g0[0] = 1u;                                    // count=1, user descriptor
  g0[1] = lds_addr;                              // lds_addr[63:32]
  g0[2] = (unsigned)ga;                          // global_addr low
  g0[3] = (unsigned)((ga >> 32) & 0x01FFFFFFu) | (2u << 30);  // addr hi | type=2
  u32x8 g1;
  g1[0] = 1u << 16;                              // data_size=1 (2 bytes), no mask
  g1[1] = (tensor_d0 & 0xFFFFu) << 16;           // tensor_dim0[15:0]
  g1[2] = (tensor_d0 >> 16) | ((tensor_d1 & 0xFFFFu) << 16);  // d0 hi | d1 lo
  g1[3] = (tensor_d1 >> 16) | (tile_d0 << 16);   // d1 hi | tile_dim0
  g1[4] = tile_d1;                               // tile_dim1 (tile_dim2=0)
  g1[5] = stride_d0;                             // tensor_dim0_stride low 32
  g1[6] = 0u;                                    // stride d0 hi | stride d1 lo
  g1[7] = 0u;
  asm volatile("tensor_load_to_lds %0, %1" :: "s"(g0), "s"(g1) : "memory");
}

// ---------- f32 -> bf16 bulk convert ----------
__global__ void convert_bf16(const float* __restrict__ src,
                             unsigned short* __restrict__ dst, int n) {
  int idx = blockIdx.x * blockDim.x + threadIdx.x;
  if (idx < n) dst[idx] = f2bf(src[idx]);
}

// ---------- weight transpose + convert: w[N][K] f32 -> wT[K][N] bf16 (N==512) ----------
__global__ void transpose_bf16(const float* __restrict__ w,
                               unsigned short* __restrict__ wt,
                               int rows /*N=512*/, int cols /*K*/) {
  int idx = blockIdx.x * blockDim.x + threadIdx.x;
  int total = rows * cols;
  if (idx >= total) return;
  int n = idx / cols;
  int k = idx % cols;
  wt[(size_t)k * rows + n] = f2bf(w[idx]);
}

// ---------- big GEMM: pre[M][512] = A[M][K](bf16) @ wT[K][512](bf16) + b0 + b1 ----------
__global__ void gemm_pre(const unsigned short* __restrict__ A,
                         const unsigned short* __restrict__ BT,
                         const float* __restrict__ b0, const float* __restrict__ b1,
                         float* __restrict__ pre, int K) {
  __shared__ unsigned short As[2][128 * 32];  // 8 KB per buffer
  __shared__ unsigned short Bs[2][32 * 64];   // 4 KB per buffer

  const int tid  = threadIdx.x;
  const int wave = tid >> 5;
  const int lane = tid & 31;
  const int wm = wave >> 1;   // 0..3
  const int wn = wave & 1;    // 0..1
  const int m0 = blockIdx.y * 128;
  const int n0 = blockIdx.x * 64;

  const unsigned as_lds = (unsigned)(size_t)&As[0][0];
  const unsigned bs_lds = (unsigned)(size_t)&Bs[0][0];

  auto issue_stage = [&](int stage, int buf) {
#pragma unroll
    for (int rep = 0; rep < 2; ++rep) {
      int ch = tid + rep * 256;
      int r = ch >> 2, c = (ch & 3) * 8;
      async_b128(as_lds + (unsigned)(buf * 8192 + ch * 16),
                 A + (size_t)(m0 + r) * K + stage * 32 + c);
    }
    {
      int r = tid >> 3, c = (tid & 7) * 8;
      async_b128(bs_lds + (unsigned)(buf * 4096 + tid * 16),
                 BT + (size_t)(stage * 32 + r) * HH + n0 + c);
    }
  };

  f32x8 acc[2][2] = {};
  const int nstages = K / 32;
  issue_stage(0, 0);

  for (int s = 0; s < nstages; ++s) {
    if (s + 1 < nstages) {
      issue_stage(s + 1, (s + 1) & 1);
      wait_async_3();
    } else {
      wait_async_0();
    }
    __syncthreads();

    const int buf = s & 1;
    const unsigned short* Ab = &As[buf][0];
    const unsigned short* Bb = &Bs[buf][0];
    const int g = lane >> 4;

    bf16x16 af[2];
#pragma unroll
    for (int sm = 0; sm < 2; ++sm) {
      int mrow = wm * 32 + sm * 16 + (lane & 15);
#pragma unroll
      for (int v = 0; v < 8; ++v) {
        int k = ((v >> 2) << 4) + ((v & 3) << 1) + (g << 3);
        af[sm][2 * v]     = us2bf(Ab[mrow * 32 + k]);
        af[sm][2 * v + 1] = us2bf(Ab[mrow * 32 + k + 1]);
      }
    }
    bf16x16 bfq[2];
#pragma unroll
    for (int sn = 0; sn < 2; ++sn) {
      int nb = wn * 32 + sn * 16;
#pragma unroll
      for (int v = 0; v < 8; ++v) {
        int k = ((v >> 2) << 4) + ((v & 3) << 1) + (g << 3);
        bfq[sn][2 * v]     = us2bf(Bb[k * 64 + nb + (lane & 15)]);
        bfq[sn][2 * v + 1] = us2bf(Bb[(k + 1) * 64 + nb + (lane & 15)]);
      }
    }
#pragma unroll
    for (int sm = 0; sm < 2; ++sm)
#pragma unroll
      for (int sn = 0; sn < 2; ++sn)
        acc[sm][sn] = wmma_bf16(af[sm], bfq[sn], acc[sm][sn]);
    __syncthreads();
  }

  const int g = lane >> 4;
#pragma unroll
  for (int sm = 0; sm < 2; ++sm)
#pragma unroll
    for (int sn = 0; sn < 2; ++sn) {
      int n = n0 + wn * 32 + sn * 16 + (lane & 15);
      float bias = b0[n] + b1[n];
#pragma unroll
      for (int v = 0; v < 8; ++v) {
        int m = m0 + wm * 32 + sm * 16 + v + 8 * g;
        pre[(size_t)m * HH + n] = acc[sm][sn][v] + bias;
      }
    }
}

// ---------- init ----------
__global__ void scan_init(unsigned short* hbuf, float* hlast, int* ctr) {
  int idx = blockIdx.x * blockDim.x + threadIdx.x;
  if (idx < 2 * 4 * 16 * HH) hbuf[idx] = 0;
  if (idx < BB * HH) hlast[idx] = 0.0f;
  if (idx < 4) ctr[idx] = 0;
}

// ---------- recurrent scan ----------
// grid (8 n-chunks, 4 batch slabs), 128 threads (4 waves). W_hh^T slice (64 KB)
// arrives via one TDM tensor_load_to_lds; each step async-DMAs the 16 KB h
// broadcast, computes 4x (16x16 x K=512) WMMA tiles, v_tanh_f32, and syncs the
// 8 WGs of the slab with a release/acquire atomic barrier (double-buffered h).
__global__ void rnn_scan(const unsigned short* __restrict__ WT,
                         const float* __restrict__ pre,
                         unsigned short* __restrict__ hbuf,
                         float* __restrict__ hlast,
                         unsigned short* __restrict__ outbf,
                         int* __restrict__ ctr) {
  extern __shared__ unsigned short smem[];
  unsigned short* Ws = smem;              // [512][64] (64 KB)
  unsigned short* Hs = smem + 512 * 64;   // [16][512] (16 KB)

  const int tid  = threadIdx.x;
  const int wave = tid >> 5;
  const int lane = tid & 31;
  const int nc   = blockIdx.x;
  const int slab = blockIdx.y;

  const unsigned ws_lds = (unsigned)(size_t)Ws;
  const unsigned hs_lds = (unsigned)(size_t)Hs;

  // one-time W slice: TDM pulls the strided 64x512 tile into packed LDS.
  if (wave == 0) {
    tdm_load_2d(ws_lds, WT + nc * 64,
                /*tensor_d0=*/HH, /*tensor_d1=*/HH,
                /*tile_d0=*/64, /*tile_d1=*/HH, /*stride_d0=*/HH);
    __builtin_amdgcn_s_wait_tensorcnt(0);
  }
  __syncthreads();

  const int g = lane >> 4;
  const int nloc  = wave * 16 + (lane & 15);
  const int nglob = nc * 64 + nloc;

  for (int t = 0; t < SS; ++t) {
    // async-DMA current h (16 KB) into LDS
    const unsigned short* hcur =
        hbuf + (size_t)(t & 1) * (4 * 16 * HH) + (size_t)slab * 16 * HH;
#pragma unroll
    for (int i = 0; i < 8; ++i) {
      int idx = tid + i * 128;
      async_b128(hs_lds + (unsigned)(idx * 16), hcur + idx * 8);
    }

    // pre operands do not depend on h: load them now, hide under WMMA chain
    float p[8];
#pragma unroll
    for (int v = 0; v < 8; ++v) {
      int b = slab * 16 + v + 8 * g;
      p[v] = pre[((size_t)b * SS + t) * HH + nglob];
      if (t + 1 < SS)
        __builtin_prefetch(&pre[((size_t)b * SS + t + 1) * HH + nglob], 0, 1);
    }

    wait_async_0();
    __syncthreads();

    f32x8 acc = {};
#pragma unroll 4
    for (int kc = 0; kc < 16; ++kc) {
      bf16x16 af, bfq;
      int mrow = lane & 15;
#pragma unroll
      for (int v = 0; v < 8; ++v) {
        int k = kc * 32 + ((v >> 2) << 4) + ((v & 3) << 1) + (g << 3);
        af[2 * v]     = us2bf(Hs[mrow * HH + k]);
        af[2 * v + 1] = us2bf(Hs[mrow * HH + k + 1]);
      }
#pragma unroll
      for (int v = 0; v < 8; ++v) {
        int krow = kc * 32 + ((v >> 2) << 4) + ((v & 3) << 1) + (g << 3);
        bfq[2 * v]     = us2bf(Ws[krow * 64 + wave * 16 + (lane & 15)]);
        bfq[2 * v + 1] = us2bf(Ws[(krow + 1) * 64 + wave * 16 + (lane & 15)]);
      }
      acc = wmma_bf16(af, bfq, acc);
    }

    unsigned short* hnext =
        hbuf + (size_t)((t + 1) & 1) * (4 * 16 * HH) + (size_t)slab * 16 * HH;
#pragma unroll
    for (int v = 0; v < 8; ++v) {
      int m = v + 8 * g;
      int b = slab * 16 + m;
      float val = fast_tanh(acc[v] + p[v]);
      unsigned short hv = f2bf(val);
      hnext[m * HH + nglob] = hv;
      if (outbf) outbf[((size_t)b * SS + t) * HH + nglob] = hv;
      if (hlast && t == SS - 1) hlast[(size_t)b * HH + nglob] = val;
    }

    // inter-WG step barrier (8 WGs per slab)
    __threadfence();
    __syncthreads();
    if (tid == 0) {
      __hip_atomic_fetch_add(&ctr[slab], 1, __ATOMIC_RELEASE, __HIP_MEMORY_SCOPE_AGENT);
      int target = 8 * (t + 1);
      while (__hip_atomic_load(&ctr[slab], __ATOMIC_ACQUIRE, __HIP_MEMORY_SCOPE_AGENT) < target)
        __builtin_amdgcn_s_sleep(1);
    }
    __syncthreads();
    __builtin_amdgcn_s_cluster_barrier();  // NOP outside cluster dispatch
  }
}

// ---------- classification head ----------
__global__ void fc_head(const float* __restrict__ hlast, const float* __restrict__ wfc,
                        const float* __restrict__ bfc, float* __restrict__ out) {
  int tid = threadIdx.x;
  if (tid >= BB * CC) return;
  int b = tid / CC, c = tid % CC;
  float s = bfc[c];
  for (int h = 0; h < HH; ++h) s += hlast[b * HH + h] * wfc[c * HH + h];
  out[b * CC + c] = s;
}

// ---------- host ----------
extern "C" void kernel_launch(void* const* d_in, const int* in_sizes, int n_in,
                              void* d_out, int out_size, void* d_ws, size_t ws_size,
                              hipStream_t stream) {
  const float* x     = (const float*)d_in[0];
  const float* w_ih0 = (const float*)d_in[1];
  const float* w_hh0 = (const float*)d_in[2];
  const float* b_ih0 = (const float*)d_in[3];
  const float* b_hh0 = (const float*)d_in[4];
  const float* w_ih1 = (const float*)d_in[5];
  const float* w_hh1 = (const float*)d_in[6];
  const float* b_ih1 = (const float*)d_in[7];
  const float* b_hh1 = (const float*)d_in[8];
  const float* w_fc  = (const float*)d_in[9];
  const float* b_fc  = (const float*)d_in[10];
  float* out = (float*)d_out;

  char* ws = (char*)d_ws;
  size_t off = 0;
  auto carve = [&](size_t bytes) {
    void* p = ws + off;
    off = (off + bytes + 255) & ~(size_t)255;
    return p;
  };
  unsigned short* xbf    = (unsigned short*)carve((size_t)MM * II * 2);
  unsigned short* wt_ih0 = (unsigned short*)carve((size_t)II * HH * 2);
  unsigned short* wt_hh0 = (unsigned short*)carve((size_t)HH * HH * 2);
  unsigned short* wt_ih1 = (unsigned short*)carve((size_t)HH * HH * 2);
  unsigned short* wt_hh1 = (unsigned short*)carve((size_t)HH * HH * 2);
  float*          pre    = (float*)carve((size_t)MM * HH * 4);
  unsigned short* out0   = (unsigned short*)carve((size_t)MM * HH * 2);
  unsigned short* hbuf   = (unsigned short*)carve((size_t)2 * 4 * 16 * HH * 2);
  float*          hlast  = (float*)carve((size_t)BB * HH * 4);
  int*            ctr    = (int*)carve(4 * sizeof(int));
  (void)ws_size; (void)in_sizes; (void)n_in; (void)out_size;

  convert_bf16<<<(MM * II + 255) / 256, 256, 0, stream>>>(x, xbf, MM * II);
  transpose_bf16<<<(HH * II + 255) / 256, 256, 0, stream>>>(w_ih0, wt_ih0, HH, II);
  transpose_bf16<<<(HH * HH + 255) / 256, 256, 0, stream>>>(w_hh0, wt_hh0, HH, HH);
  transpose_bf16<<<(HH * HH + 255) / 256, 256, 0, stream>>>(w_ih1, wt_ih1, HH, HH);
  transpose_bf16<<<(HH * HH + 255) / 256, 256, 0, stream>>>(w_hh1, wt_hh1, HH, HH);

  dim3 ggrid(HH / 64, MM / 128);
  const size_t scan_lds = (512 * 64 + 16 * HH) * sizeof(unsigned short); // 80 KB
  dim3 sgrid(8, 4);

  gemm_pre<<<ggrid, 256, 0, stream>>>(xbf, wt_ih0, b_ih0, b_hh0, pre, II);
  scan_init<<<(2 * 4 * 16 * HH + 255) / 256, 256, 0, stream>>>(hbuf, hlast, ctr);
  rnn_scan<<<sgrid, 128, scan_lds, stream>>>(wt_hh0, pre, hbuf, nullptr, out0, ctr);

  gemm_pre<<<ggrid, 256, 0, stream>>>(out0, wt_ih1, b_ih1, b_hh1, pre, HH);
  scan_init<<<(2 * 4 * 16 * HH + 255) / 256, 256, 0, stream>>>(hbuf, hlast, ctr);
  rnn_scan<<<sgrid, 128, scan_lds, stream>>>(wt_hh1, pre, hbuf, hlast, nullptr, ctr);

  fc_head<<<1, BB * CC, 0, stream>>>(hlast, w_fc, b_fc, out);
}
